// MolGraph_Encoder_51943334478095
// MI455X (gfx1250) — matrix operand order repaired and checked by reading, not compiled, not run on hardware
//
#include <hip/hip_runtime.h>
#include <hip/hip_bf16.h>
#include <math.h>

typedef __bf16 bf16_t;
typedef bf16_t v16bf __attribute__((ext_vector_type(16)));
typedef float  v8f   __attribute__((ext_vector_type(8)));

#define DEV __device__ __forceinline__

DEV unsigned fkey_encode(float f) {
  unsigned u = __float_as_uint(f);
  return (u & 0x80000000u) ? ~u : (u | 0x80000000u);
}
DEV float fkey_decode(unsigned k) {
  unsigned u = (k & 0x80000000u) ? (k & 0x7FFFFFFFu) : ~k;
  return __uint_as_float(u);
}
DEV float leakyf(float x) { return x > 0.f ? x : 0.01f * x; }
DEV float sigmoidf(float x) { return 1.f / (1.f + __expf(-x)); }

// ---------------------------------------------------------------------------
// WMMA GEMM: C[M,Nn] = act(A[M,K](bf16,rm) x Bt[Nn,K](bf16, K-major) + bias)
// block = 256 thr = 8 waves; block tile 128x64; wave tile 32x32 (2x2 WMMA)
// Both A and B fragments load as contiguous 2x ds_load_b128 per fragment.
// ---------------------------------------------------------------------------
__global__ __launch_bounds__(256)
void wmma_gemm_bf16(const bf16_t* __restrict__ A, const bf16_t* __restrict__ Bt,
                    const float* __restrict__ bias,
                    float* __restrict__ C, bf16_t* __restrict__ Cbf,
                    int M, int Nn, int K, int act)
{
  __shared__ __align__(16) bf16_t As[128][40];   // 128 rows(m) x 32(k), pad 40
  __shared__ __align__(16) bf16_t Bst[64][40];   // 64 rows(n)  x 32(k), pad 40
  const int tid  = threadIdx.x;
  const int lane = tid & 31, wid = tid >> 5;
  const int wm = (wid & 3) * 32, wn = (wid >> 2) * 32;
  const int lm = lane & 15, half = lane >> 4;
  const int m0 = blockIdx.y * 128, n0 = blockIdx.x * 64;

  v8f acc[2][2] = {};

  for (int kk = 0; kk < K; kk += 32) {
    // stage A tile 128x32 (bounds-checked rows, zero fill)
#pragma unroll
    for (int i = tid; i < 128 * 4; i += 256) {
      int r = i >> 2, ch = i & 3;
      int rg = m0 + r;
      uint4 v = make_uint4(0u, 0u, 0u, 0u);
      if (rg < M) v = *(const uint4*)(A + (size_t)rg * K + kk + ch * 8);
      *(uint4*)(&As[r][ch * 8]) = v;
    }
    // stage B tile 64(n) x 32(k): one uint4 per thread, fully coalesced
    {
      int r = tid >> 2, ch = tid & 3;
      uint4 v = *(const uint4*)(Bt + (size_t)(n0 + r) * K + kk + ch * 8);
      *(uint4*)(&Bst[r][ch * 8]) = v;
    }
    __syncthreads();

    v16bf afrag[2], bfrag[2];
#pragma unroll
    for (int mi = 0; mi < 2; ++mi) {
#pragma unroll
      for (int i = 0; i < 8; ++i)
        afrag[mi][i] = As[wm + mi * 16 + lm][half * 8 + i];
#pragma unroll
      for (int i = 0; i < 8; ++i)
        afrag[mi][8 + i] = As[wm + mi * 16 + lm][16 + half * 8 + i];
    }
#pragma unroll
    for (int ni = 0; ni < 2; ++ni)
#pragma unroll
      for (int i = 0; i < 16; ++i)
        bfrag[ni][i] = Bst[wn + ni * 16 + lm][(i >> 3) * 16 + half * 8 + (i & 7)];

#pragma unroll
    for (int mi = 0; mi < 2; ++mi)
#pragma unroll
      for (int ni = 0; ni < 2; ++ni)
        acc[mi][ni] = __builtin_amdgcn_wmma_f32_16x16x32_bf16(
            false, afrag[mi], false, bfrag[ni], (short)0, acc[mi][ni], false, false);
    __syncthreads();
  }

#pragma unroll
  for (int mi = 0; mi < 2; ++mi)
#pragma unroll
    for (int ni = 0; ni < 2; ++ni) {
      int col = n0 + wn + ni * 16 + lm;
      float bv = bias ? bias[col] : 0.f;
#pragma unroll
      for (int r = 0; r < 8; ++r) {
        int row = m0 + wm + mi * 16 + half * 8 + r;
        if (row < M) {
          float v = acc[mi][ni][r] + bv;
          if (act == 1) v = leakyf(v);
          else if (act == 2) v = fmaxf(v, 0.f);
          C[(size_t)row * Nn + col] = v;
          if (Cbf) Cbf[(size_t)row * Nn + col] = (bf16_t)v;
        }
      }
    }
}

// ---------------------------------------------------------------------------
// conversion / utility kernels
// ---------------------------------------------------------------------------
__global__ void fill_f32_kernel(float* p, float v, long n) {
  long i = (long)blockIdx.x * blockDim.x + threadIdx.x;
  if (i < n) p[i] = v;
}

// straight f32 -> bf16 convert (used for W.T operands already K-major)
__global__ void cvt_kernel(const float* src, bf16_t* dst, long n) {
  long i = (long)blockIdx.x * blockDim.x + threadIdx.x;
  if (i < n) dst[i] = (bf16_t)src[i];
}

// W[K,Nn] f32 row-major -> Wt[Nn,Kpad] bf16 (transposed B operand, K padded)
__global__ void cvt_wt_kernel(const float* src, bf16_t* dst, int K, int Kpad, int Nn) {
  long idx = (long)blockIdx.x * blockDim.x + threadIdx.x;
  if (idx >= (long)Nn * Kpad) return;
  int n = (int)(idx / Kpad), k = (int)(idx % Kpad);
  dst[idx] = (k < K) ? (bf16_t)src[(long)k * Nn + n] : (bf16_t)0.f;
}

// src[rows,cols] f32 -> dst[rows,colsPad] bf16 (zero-padded columns; A operand)
__global__ void cvt_pad_cols_kernel(const float* src, bf16_t* dst, long rows, int cols, int colsPad) {
  long idx = (long)blockIdx.x * blockDim.x + threadIdx.x;
  if (idx >= rows * colsPad) return;
  long r = idx / colsPad; int c = (int)(idx % colsPad);
  dst[idx] = (c < cols) ? (bf16_t)src[r * cols + c] : (bf16_t)0.f;
}

// dst[e, 0:74]=atom[src[e]], [74:86]=bond[e], [86:96]=0   (bf16, K padded to 96)
__global__ void build_edge_feat_kernel(const float* atom, const float* bond, const int* srci,
                                       bf16_t* dst, int E) {
  long idx = (long)blockIdx.x * blockDim.x + threadIdx.x;
  if (idx >= (long)E * 96) return;
  int e = (int)(idx / 96), k = (int)(idx % 96);
  float v = 0.f;
  if (k < 74)      v = atom[(long)srci[e] * 74 + k];
  else if (k < 86) v = bond[(long)e * 12 + (k - 74)];
  dst[idx] = (bf16_t)v;
}

// per-row dot: out1[r]=X[r,:128].w1 (+out2=X.w2); one wave32 per row
__global__ void rowdot2_kernel(const float* __restrict__ X, long rows,
                               const float* __restrict__ w1, const float* __restrict__ w2,
                               float* __restrict__ out1, float* __restrict__ out2) {
  long row = (long)blockIdx.x * (blockDim.x >> 5) + (threadIdx.x >> 5);
  int lane = threadIdx.x & 31;
  if (row >= rows) return;
  const float* xr = X + row * 128;
  float s1 = 0.f, s2 = 0.f;
#pragma unroll
  for (int i = 0; i < 4; ++i) {
    float xv = xr[lane + 32 * i];
    s1 += xv * w1[lane + 32 * i];
    if (w2) s2 += xv * w2[lane + 32 * i];
  }
#pragma unroll
  for (int off = 16; off > 0; off >>= 1) {
    s1 += __shfl_down(s1, off, 32);
    if (w2) s2 += __shfl_down(s2, off, 32);
  }
  if (lane == 0) { out1[row] = s1; if (w2) out2[row] = s2; }
}

// GetContext logits: lg = leaky(s1[dst] + s2[e] + b); seg-max via monotone key
__global__ void edge_logit_ctx_kernel(const float* s1, const float* s2, const int* dsti,
                                      const float* bptr, float* lg, unsigned* segkey, int E) {
  int e = blockIdx.x * blockDim.x + threadIdx.x;
  if (e >= E) return;
  float v = leakyf(s1[dsti[e]] + s2[e] + bptr[0]);
  lg[e] = v;
  atomicMax(&segkey[dsti[e]], fkey_encode(v));
}

// Layer logits: lg = leaky(p1[dst] + p2[src] + b)
__global__ void edge_logit_layer_kernel(const float* p1, const float* p2, const int* srci,
                                        const int* dsti, const float* bptr, float* lg,
                                        unsigned* segkey, int E) {
  int e = blockIdx.x * blockDim.x + threadIdx.x;
  if (e >= E) return;
  float v = leakyf(p1[dsti[e]] + p2[srci[e]] + bptr[0]);
  lg[e] = v;
  atomicMax(&segkey[dsti[e]], fkey_encode(v));
}

__global__ void edge_exp_kernel(const float* lg, const unsigned* segkey, const int* dsti,
                                float* aun, float* segsum, int E) {
  int e = blockIdx.x * blockDim.x + threadIdx.x;
  if (e >= E) return;
  int d = dsti[e];
  float a = __expf(lg[e] - fkey_decode(segkey[d]));
  aun[e] = a;
  atomicAdd(&segsum[d], a);
}

__global__ void edge_norm_kernel(float* aun, const float* segsum, const int* dsti, int E) {
  int e = blockIdx.x * blockDim.x + threadIdx.x;
  if (e >= E) return;
  aun[e] = aun[e] / segsum[dsti[e]];
}

// GetContext scatter: c[dst] += a[e] * et[e,:]
__global__ void scatter_ctx_kernel(const float* a, const float* et, const int* dsti,
                                   float* c, int E) {
  long idx = (long)blockIdx.x * blockDim.x + threadIdx.x;
  if (idx >= (long)E * 128) return;
  int e = (int)(idx >> 7), d = (int)(idx & 127);
  atomicAdd(&c[(long)dsti[e] * 128 + d], a[e] * et[idx]);
}

// Layer scatter: c[dst] += a[e] * hp[src,:]
__global__ void scatter_layer_kernel(const float* a, const float* hp, const int* srci,
                                     const int* dsti, float* c, int E) {
  long idx = (long)blockIdx.x * blockDim.x + threadIdx.x;
  if (idx >= (long)E * 128) return;
  int e = (int)(idx >> 7), d = (int)(idx & 127);
  atomicAdd(&c[(long)dsti[e] * 128 + d], a[e] * hp[(long)srci[e] * 128 + d]);
}

// x = elu(c) -> bf16
__global__ void elu_bf16_kernel(const float* c, bf16_t* x, long n) {
  long i = (long)blockIdx.x * blockDim.x + threadIdx.x;
  if (i >= n) return;
  float v = c[i];
  v = v > 0.f ? v : (__expf(v) - 1.f);
  x[i] = (bf16_t)v;
}

// fused GRU gates + relu; updates h (f32) and h_bf (bf16) in place
__global__ void gru_gate_kernel(const float* __restrict__ gi, const float* __restrict__ gh,
                                float* __restrict__ h, bf16_t* __restrict__ hbf, long Nn) {
  long idx = (long)blockIdx.x * blockDim.x + threadIdx.x;
  if (idx >= Nn * 128) return;
  long n = idx >> 7; int d = (int)(idx & 127);
  const float* gin = gi + n * 384;
  const float* ghn = gh + n * 384;
  float ir = gin[d], iz = gin[128 + d], ig = gin[256 + d];
  float hr = ghn[d], hz = ghn[128 + d], hg = ghn[256 + d];
  float hv = h[idx];
  float r = sigmoidf(ir + hr);
  float z = sigmoidf(iz + hz);
  float ng = tanhf(ig + r * hg);
  float hn = fmaxf((1.f - z) * ng + z * hv, 0.f);
  h[idx] = hn;
  hbf[idx] = (bf16_t)hn;
}

__global__ void readout_scatter_kernel(const float* t2, const int* gid, float* out, long Nn) {
  long idx = (long)blockIdx.x * blockDim.x + threadIdx.x;
  if (idx >= Nn * 128) return;
  long n = idx >> 7; int d = (int)(idx & 127);
  atomicAdd(&out[(long)gid[n] * 128 + d], t2[idx]);
}

__global__ void count_kernel(const int* gid, float* counts, int Nn) {
  int n = blockIdx.x * blockDim.x + threadIdx.x;
  if (n >= Nn) return;
  atomicAdd(&counts[gid[n]], 1.f);
}

__global__ void divide_kernel(float* out, const float* counts, int Gn) {
  int idx = blockIdx.x * blockDim.x + threadIdx.x;
  if (idx >= Gn * 128) return;
  out[idx] = out[idx] / fmaxf(counts[idx >> 7], 1.f);
}

// ---------------------------------------------------------------------------
// host orchestration
// ---------------------------------------------------------------------------
static inline long cdivl(long a, long b) { return (a + b - 1) / b; }

extern "C" void kernel_launch(void* const* d_in, const int* in_sizes, int n_in,
                              void* d_out, int out_size, void* d_ws, size_t ws_size,
                              hipStream_t stream) {
  const int N = 100000, E = 400000, G = 4000, D = 128, L = 11;
  const int KA = 96;  // padded K for atom (74) and atom+bond (86) features

  // ---- inputs (setup_inputs order) ----
  const float* atom_feats = (const float*)d_in[0];
  const float* bond_feats = (const float*)d_in[1];
  const int*   srci       = (const int*)d_in[2];
  const int*   dsti       = (const int*)d_in[3];
  const int*   gids       = (const int*)d_in[4];
  const float* W_node  = (const float*)d_in[5];
  const float* b_node  = (const float*)d_in[6];
  const float* W_edge1 = (const float*)d_in[7];
  const float* b_edge1 = (const float*)d_in[8];
  const float* W_edge2 = (const float*)d_in[9];
  const float* b_edge2 = (const float*)d_in[10];
  const float* W_et    = (const float*)d_in[11];
  const float* b_et    = (const float*)d_in[12];
  const float* g0_Wih  = (const float*)d_in[13];
  const float* g0_bih  = (const float*)d_in[14];
  const float* g0_Whh  = (const float*)d_in[15];
  const float* g0_bhh  = (const float*)d_in[16];
  const float* Wpe     = (const float*)d_in[17];
  const float* bpe     = (const float*)d_in[18];
  const float* Wpn     = (const float*)d_in[19];
  const float* bpn     = (const float*)d_in[20];
  const float* gWih    = (const float*)d_in[21];
  const float* gbih    = (const float*)d_in[22];
  const float* gWhh    = (const float*)d_in[23];
  const float* gbhh    = (const float*)d_in[24];
  const float* Wr1     = (const float*)d_in[25];
  const float* br1     = (const float*)d_in[26];
  const float* Wr2     = (const float*)d_in[27];
  const float* br2     = (const float*)d_in[28];
  float* out = (float*)d_out;

  // ---- workspace bump allocator ----
  char* wp = (char*)d_ws;
  auto alloc = [&](size_t bytes) -> void* {
    void* r = (void*)wp;
    wp += (bytes + 255) & ~(size_t)255;
    return r;
  };
  bf16_t* atom_bf  = (bf16_t*)alloc((size_t)N * KA * 2);
  bf16_t* edge_bf  = (bf16_t*)alloc((size_t)E * KA * 2);
  bf16_t* Wnode_bt = (bf16_t*)alloc((size_t)D * KA * 2);      // [128][96]
  bf16_t* We1_bt   = (bf16_t*)alloc((size_t)D * KA * 2);      // [128][96]
  bf16_t* Wet_bt   = (bf16_t*)alloc((size_t)D * D * 2);       // [128][128]
  bf16_t* Wr1_bt   = (bf16_t*)alloc((size_t)D * D * 2);
  bf16_t* Wr2_bt   = (bf16_t*)alloc((size_t)D * D * 2);
  bf16_t* Wpn_bt   = (bf16_t*)alloc((size_t)L * D * D * 2);
  bf16_t* Wih0_bt  = (bf16_t*)alloc((size_t)3 * D * D * 2);   // [384][128] = Wih as-is
  bf16_t* Whh0_bt  = (bf16_t*)alloc((size_t)3 * D * D * 2);
  bf16_t* Wih_bt   = (bf16_t*)alloc((size_t)L * 3 * D * D * 2);
  bf16_t* Whh_bt   = (bf16_t*)alloc((size_t)L * 3 * D * D * 2);
  float*  h     = (float*)alloc((size_t)N * D * 4);
  bf16_t* h_bf  = (bf16_t*)alloc((size_t)N * D * 2);
  float*  he1   = (float*)alloc((size_t)E * D * 4);
  bf16_t* he1bf = (bf16_t*)alloc((size_t)E * D * 2);
  float*  et    = (float*)alloc((size_t)E * D * 4);
  float*  cbuf  = (float*)alloc((size_t)N * D * 4);
  bf16_t* x_bf  = (bf16_t*)alloc((size_t)N * D * 2);
  float*  gi    = (float*)alloc((size_t)N * 3 * D * 4);
  float*  gh    = (float*)alloc((size_t)N * 3 * D * 4);
  float*  hp    = (float*)alloc((size_t)N * D * 4);
  bf16_t* t1_bf = (bf16_t*)alloc((size_t)N * D * 2);
  float*  p1    = (float*)alloc((size_t)N * 4);
  float*  p2    = (float*)alloc((size_t)N * 4);
  float*  s2    = (float*)alloc((size_t)E * 4);
  float*  lg    = (float*)alloc((size_t)E * 4);
  float*  aun   = (float*)alloc((size_t)E * 4);
  unsigned* segkey = (unsigned*)alloc((size_t)N * 4);
  float*  segsum   = (float*)alloc((size_t)N * 4);
  float*  counts   = (float*)alloc((size_t)G * 4);

  const int T = 256;
  auto gemm = [&](const bf16_t* A, const bf16_t* Bt, const float* bias, float* C,
                  bf16_t* Cbf, int M, int Nn, int K, int act) {
    dim3 grid(Nn / 64, (unsigned)cdivl(M, 128));
    wmma_gemm_bf16<<<grid, 256, 0, stream>>>(A, Bt, bias, C, Cbf, M, Nn, K, act);
  };
  auto fill = [&](float* p, float v, long n) {
    fill_f32_kernel<<<(unsigned)cdivl(n, T), T, 0, stream>>>(p, v, n);
  };

  // ---- weight conversions to bf16, all stored K-major (transposed B) ----
  cvt_wt_kernel<<<(unsigned)cdivl((long)D * KA, T), T, 0, stream>>>(W_node, Wnode_bt, 74, KA, D);
  cvt_wt_kernel<<<(unsigned)cdivl((long)D * KA, T), T, 0, stream>>>(W_edge1, We1_bt, 86, KA, D);
  cvt_wt_kernel<<<(unsigned)cdivl((long)D * D, T), T, 0, stream>>>(W_et, Wet_bt, D, D, D);
  cvt_wt_kernel<<<(unsigned)cdivl((long)D * D, T), T, 0, stream>>>(Wr1, Wr1_bt, D, D, D);
  cvt_wt_kernel<<<(unsigned)cdivl((long)D * D, T), T, 0, stream>>>(Wr2, Wr2_bt, D, D, D);
  // GRU weights: x @ W.T  =>  Bt = W itself ([3D,D], K-major already). Straight convert.
  cvt_kernel<<<(unsigned)cdivl((long)3 * D * D, T), T, 0, stream>>>(g0_Wih, Wih0_bt, (long)3 * D * D);
  cvt_kernel<<<(unsigned)cdivl((long)3 * D * D, T), T, 0, stream>>>(g0_Whh, Whh0_bt, (long)3 * D * D);
  cvt_kernel<<<(unsigned)cdivl((long)L * 3 * D * D, T), T, 0, stream>>>(gWih, Wih_bt, (long)L * 3 * D * D);
  cvt_kernel<<<(unsigned)cdivl((long)L * 3 * D * D, T), T, 0, stream>>>(gWhh, Whh_bt, (long)L * 3 * D * D);
  for (int l = 0; l < L; ++l)
    cvt_wt_kernel<<<(unsigned)cdivl((long)D * D, T), T, 0, stream>>>(
        Wpn + (size_t)l * D * D, Wpn_bt + (size_t)l * D * D, D, D, D);

  // ---- activations to bf16 ----
  cvt_pad_cols_kernel<<<(unsigned)cdivl((long)N * KA, T), T, 0, stream>>>(atom_feats, atom_bf, N, 74, KA);
  build_edge_feat_kernel<<<(unsigned)cdivl((long)E * KA, T), T, 0, stream>>>(atom_feats, bond_feats, srci, edge_bf, E);

  // ======================= GetContext =======================
  gemm(atom_bf, Wnode_bt, b_node, h, h_bf, N, D, KA, /*leaky*/1);        // hv_new
  gemm(edge_bf, We1_bt, b_edge1, he1, he1bf, E, D, KA, /*leaky*/1);      // he1

  rowdot2_kernel<<<(unsigned)cdivl(N, 8), 256, 0, stream>>>(h, N, W_edge2, nullptr, p1, nullptr);
  rowdot2_kernel<<<(unsigned)cdivl(E, 8), 256, 0, stream>>>(he1, E, W_edge2 + D, nullptr, s2, nullptr);

  fill((float*)segkey, 0.f, N);  // key 0 == minimal encoding
  fill(segsum, 0.f, N);
  fill(cbuf, 0.f, (long)N * D);
  edge_logit_ctx_kernel<<<(unsigned)cdivl(E, T), T, 0, stream>>>(p1, s2, dsti, b_edge2, lg, segkey, E);
  edge_exp_kernel<<<(unsigned)cdivl(E, T), T, 0, stream>>>(lg, segkey, dsti, aun, segsum, E);
  edge_norm_kernel<<<(unsigned)cdivl(E, T), T, 0, stream>>>(aun, segsum, dsti, E);

  gemm(he1bf, Wet_bt, b_et, et, nullptr, E, D, D, /*none*/0);            // he1 @ W_et
  scatter_ctx_kernel<<<(unsigned)cdivl((long)E * D, T), T, 0, stream>>>(aun, et, dsti, cbuf, E);

  elu_bf16_kernel<<<(unsigned)cdivl((long)N * D, T), T, 0, stream>>>(cbuf, x_bf, (long)N * D);
  gemm(x_bf, Wih0_bt, g0_bih, gi, nullptr, N, 3 * D, D, 0);
  gemm(h_bf, Whh0_bt, g0_bhh, gh, nullptr, N, 3 * D, D, 0);
  gru_gate_kernel<<<(unsigned)cdivl((long)N * D, T), T, 0, stream>>>(gi, gh, h, h_bf, N);

  // ======================= 11 GNN layers =======================
  for (int l = 0; l < L; ++l) {
    const float* Wpe_l = Wpe + (size_t)l * 2 * D;
    rowdot2_kernel<<<(unsigned)cdivl(N, 8), 256, 0, stream>>>(h, N, Wpe_l, Wpe_l + D, p1, p2);

    fill((float*)segkey, 0.f, N);
    fill(segsum, 0.f, N);
    fill(cbuf, 0.f, (long)N * D);
    edge_logit_layer_kernel<<<(unsigned)cdivl(E, T), T, 0, stream>>>(p1, p2, srci, dsti, bpe + l, lg, segkey, E);
    edge_exp_kernel<<<(unsigned)cdivl(E, T), T, 0, stream>>>(lg, segkey, dsti, aun, segsum, E);
    edge_norm_kernel<<<(unsigned)cdivl(E, T), T, 0, stream>>>(aun, segsum, dsti, E);

    gemm(h_bf, Wpn_bt + (size_t)l * D * D, bpn + (size_t)l * D, hp, nullptr, N, D, D, 0);
    scatter_layer_kernel<<<(unsigned)cdivl((long)E * D, T), T, 0, stream>>>(aun, hp, srci, dsti, cbuf, E);

    elu_bf16_kernel<<<(unsigned)cdivl((long)N * D, T), T, 0, stream>>>(cbuf, x_bf, (long)N * D);
    gemm(x_bf, Wih_bt + (size_t)l * 3 * D * D, gbih + (size_t)l * 3 * D, gi, nullptr, N, 3 * D, D, 0);
    gemm(h_bf, Whh_bt + (size_t)l * 3 * D * D, gbhh + (size_t)l * 3 * D, gh, nullptr, N, 3 * D, D, 0);
    gru_gate_kernel<<<(unsigned)cdivl((long)N * D, T), T, 0, stream>>>(gi, gh, h, h_bf, N);
  }

  // ======================= readout (mean per graph) =======================
  gemm(h_bf, Wr1_bt, br1, gi /*scratch*/, t1_bf, N, D, D, /*relu*/2);
  gemm(t1_bf, Wr2_bt, br2, hp /*t2*/, nullptr, N, D, D, 0);

  fill(out, 0.f, (long)G * D);
  fill(counts, 0.f, G);
  readout_scatter_kernel<<<(unsigned)cdivl((long)N * D, T), T, 0, stream>>>(hp, gids, out, N);
  count_kernel<<<(unsigned)cdivl(N, T), T, 0, stream>>>(gids, counts, N);
  divide_kernel<<<(unsigned)cdivl((long)G * D, T), T, 0, stream>>>(out, counts, G);
}
